// LSTMWithoutLinear_84198538871466
// MI455X (gfx1250) — matrix operand order repaired and checked by reading, not compiled, not run on hardware
//
#include <hip/hip_runtime.h>
#include <hip/hip_bf16.h>

// ---------------------------------------------------------------------------
// LSTM seq2seq (3 layers, H=1024, D=8, B=64, 96 enc + 335 dec steps) on
// CDNA5/gfx1250.  Persistent kernel + software grid barrier; bf16 WMMA
// (v_wmma_f32_16x16x32_bf16) for all K=1024 GEMMs; cell state kept in VGPRs.
// Layer-2 fused inside block 0 via LDS (3 grid barriers/step).  K-loops use
// explicit ping/pong double buffering, two chunks per rolled iteration:
// no rotation moves, bounded register pressure (no spills), loads overlap
// the previous chunk's WMMA group.
// ---------------------------------------------------------------------------

typedef __attribute__((ext_vector_type(16))) __bf16 v16bf;
typedef __attribute__((ext_vector_type(8)))  __bf16 v8bf;
typedef __attribute__((ext_vector_type(8)))  float  v8f;

#define NBLK   32
#define TPB    256          // 8 waves per block -> 256 waves total
#define WPB    (TPB/32)
#define Hs     1024
#define Bs     64
#define Ds     8
#define ENCs   96
#define HORs   336
#define TSTEPS 431          // 96 encoder + 335 decoder steps
#define KCN    32           // H / 32 k-chunks per WMMA accumulation
#define PRED_N (Bs*HORs*Ds) // 172032
#define GSTR   ((size_t)64 * KCN * 512)   // gate stride in packed weights (elems)

__device__ __forceinline__ float sigf(float x)  { return 1.0f / (1.0f + __expf(-x)); }
__device__ __forceinline__ float tanhff(float x){ return 2.0f / (1.0f + __expf(-2.0f * x)) - 1.0f; }

// A-fragment (16x32 bf16, row-major source): lane L: m = L&15, K = 8*(L>>4)+j
// (j<8) and +j+8 (j>=8)  ->  two 16B loads at p and p+16 elements.
__device__ __forceinline__ v16bf fragA(const __bf16* p) {
    v8bf lo = *(const v8bf*)(p);
    v8bf hi = *(const v8bf*)(p + 16);
    return __builtin_shufflevector(lo, hi, 0,1,2,3,4,5,6,7,8,9,10,11,12,13,14,15);
}
// B-fragment: pre-packed lane-major, 16 contiguous bf16 per lane.
__device__ __forceinline__ v16bf fragB(const __bf16* p) {
    v8bf lo = *(const v8bf*)(p);
    v8bf hi = *(const v8bf*)(p + 8);
    return __builtin_shufflevector(lo, hi, 0,1,2,3,4,5,6,7,8,9,10,11,12,13,14,15);
}

__device__ __forceinline__ v8f wmma_bf(v16bf a, v16bf b, v8f c) {
    return __builtin_amdgcn_wmma_f32_16x16x32_bf16(false, a, false, b, (short)0, c, false, false);
}

// Sense-reversing grid barrier (all NBLK blocks resident).
__device__ __forceinline__ void gbar(unsigned* bar) {
    __threadfence();
    __syncthreads();
    if (threadIdx.x == 0) {
        unsigned gen  = __hip_atomic_load(&bar[1], __ATOMIC_RELAXED, __HIP_MEMORY_SCOPE_AGENT);
        unsigned prev = __hip_atomic_fetch_add(&bar[0], 1u, __ATOMIC_ACQ_REL, __HIP_MEMORY_SCOPE_AGENT);
        if (prev == (unsigned)(NBLK - 1)) {
            __hip_atomic_store(&bar[0], 0u, __ATOMIC_RELAXED, __HIP_MEMORY_SCOPE_AGENT);
            __hip_atomic_store(&bar[1], gen + 1u, __ATOMIC_RELEASE, __HIP_MEMORY_SCOPE_AGENT);
        } else {
            while (__hip_atomic_load(&bar[1], __ATOMIC_ACQUIRE, __HIP_MEMORY_SCOPE_AGENT) == gen)
                __builtin_amdgcn_s_sleep(1);
        }
    }
    __syncthreads();
    __threadfence();
}

// Pack W (ncols x K f32, row-major) into per-lane WMMA B fragments (bf16):
// dst[((ct*KC + kc)*32 + lane)*16 + j] = W[(ct*16 + (lane&15))*K + kc*32 + (lane>>4)*8 + (j&7) + (j>>3)*16]
__global__ void pack_weights(const float* __restrict__ W, __bf16* __restrict__ dst,
                             int nct, int K) {
    size_t idx   = (size_t)blockIdx.x * blockDim.x + threadIdx.x;
    size_t total = (size_t)nct * (size_t)(K / 32) * 512u;
    if (idx >= total) return;
    int    j    = (int)(idx & 15);
    int    lane = (int)((idx >> 4) & 31);
    size_t tile = idx >> 9;
    int    kc   = (int)(tile % (size_t)(K / 32));
    int    ct   = (int)(tile / (size_t)(K / 32));
    int    n    = ct * 16 + (lane & 15);
    int    k    = kc * 32 + (lane >> 4) * 8 + (j & 7) + ((j >> 3) << 4);
    dst[idx] = (__bf16)W[(size_t)n * K + k];
}

__global__ __launch_bounds__(TPB) void lstm_persistent(
    const float* __restrict__ inp,                           // (64, 432, 8)
    const float* __restrict__ wih0, const float* __restrict__ b0v,
    const float* __restrict__ b1v,
    const float* __restrict__ whh2, const float* __restrict__ b2v,
    const __bf16* __restrict__ pWhh0, const __bf16* __restrict__ pWih1,
    const __bf16* __restrict__ pWhh1, const __bf16* __restrict__ pWih2,
    __bf16* __restrict__ h0b,   // 2 x (64x1024) bf16 ping-pong
    __bf16* __restrict__ h1b,   // 2 x (64x1024) bf16 ping-pong
    float*  __restrict__ hdbuf, // 64x8 f32 (layer-2 hidden / decoder feedback)
    unsigned* __restrict__ bar,
    float* __restrict__ outp)   // pred base (64, 336, 8)
{
    __shared__ float g2s[Bs * 32];      // layer-2 gate preacts (block 0 only)

    const int lane = threadIdx.x & 31;
    const int nn   = lane & 15;
    const int hi   = lane >> 4;
    const int wv   = threadIdx.x >> 5;                     // wave in block, 0..7
    const int w    = blockIdx.x * WPB + wv;                // 0..255
    const int rt   = w & 3;      // batch row tile (16 rows)
    const int ut   = w >> 2;     // hidden unit tile (16 units), 0..63

    // ---- hoist step-invariant weights/biases into registers -------------
    float w0r[4][Ds], bb0[4], bb1[4];
#pragma unroll
    for (int g = 0; g < 4; ++g) {
        const int n = g * Hs + ut * 16 + nn;
        bb0[g] = b0v[n];
        bb1[g] = b1v[n];
#pragma unroll
        for (int k = 0; k < Ds; ++k) w0r[g][k] = wih0[(size_t)n * Ds + k];
    }
    // layer-2 activation constants (used by wave 0 of block 0; u = lane&7)
    const int u2 = lane & 7;
    float w2i[Ds], w2f[Ds], w2g[Ds], w2o[Ds];
#pragma unroll
    for (int k = 0; k < Ds; ++k) {
        w2i[k] = whh2[(u2)      * Ds + k];
        w2f[k] = whh2[(u2 + 8)  * Ds + k];
        w2g[k] = whh2[(u2 + 16) * Ds + k];
        w2o[k] = whh2[(u2 + 24) * Ds + k];
    }
    const float b2i = b2v[u2], b2f = b2v[u2 + 8], b2g = b2v[u2 + 16], b2o = b2v[u2 + 24];

    // per-wave packed-weight bases (lane folded in)
    const __bf16* wb0  = pWhh0 + (size_t)ut * KCN * 512 + lane * 16;
    const __bf16* wb1i = pWih1 + (size_t)ut * KCN * 512 + lane * 16;
    const __bf16* wb1h = pWhh1 + (size_t)ut * KCN * 512 + lane * 16;
    const __bf16* wb2  = pWih2 + lane * 16;

    v8f c0reg = {0,0,0,0,0,0,0,0};   // layer-0 cell state, resident in VGPRs
    v8f c1reg = {0,0,0,0,0,0,0,0};   // layer-1 cell state
    float c2reg[16];
#pragma unroll
    for (int i = 0; i < 16; ++i) c2reg[i] = 0.0f;   // layer-2 cell (wave 0 only)

    for (int t = 0; t < TSTEPS; ++t) {
        const int rbuf = t & 1, wbuf = rbuf ^ 1;
        const __bf16* h0R = h0b + (size_t)rbuf * (Bs * Hs);
        __bf16*       h0W = h0b + (size_t)wbuf * (Bs * Hs);
        const __bf16* h1R = h1b + (size_t)rbuf * (Bs * Hs);
        __bf16*       h1W = h1b + (size_t)wbuf * (Bs * Hs);
        const bool    enc   = (t < ENCs);
        const float*  xbase = enc ? (inp + (size_t)t * Ds) : hdbuf;
        const size_t  xstr  = enc ? (size_t)(ENCs + HORs) * Ds : (size_t)Ds;

        // ----------------- layer 0: g = x@Wih0^T + h@Whh0^T + b0 -----------------
        {
            v8f acc[4];
#pragma unroll
            for (int v = 0; v < 8; ++v) {
                const int m = rt * 16 + v + 8 * hi;
                const float* xr = xbase + (size_t)m * xstr;
                float x0 = xr[0], x1 = xr[1], x2 = xr[2], x3 = xr[3];
                float x4 = xr[4], x5 = xr[5], x6 = xr[6], x7 = xr[7];
#pragma unroll
                for (int g = 0; g < 4; ++g) {
                    float s = bb0[g];
                    s += x0 * w0r[g][0] + x1 * w0r[g][1] + x2 * w0r[g][2] + x3 * w0r[g][3];
                    s += x4 * w0r[g][4] + x5 * w0r[g][5] + x6 * w0r[g][6] + x7 * w0r[g][7];
                    acc[g][v] = s;
                }
            }
            const __bf16* arow = h0R + (size_t)(rt * 16 + nn) * Hs + hi * 8;
            // ping/pong pipelined k-loop, two chunks per iteration
            v16bf aA = fragA(arow);
            v16bf bA[4], bB[4];
            v16bf aB;
#pragma unroll
            for (int g = 0; g < 4; ++g) bA[g] = fragB(wb0 + g * GSTR);
#pragma unroll 1
            for (int kc = 0; kc < KCN - 2; kc += 2) {
                aB = fragA(arow + (kc + 1) * 32);
#pragma unroll
                for (int g = 0; g < 4; ++g) bB[g] = fragB(wb0 + (size_t)(kc + 1) * 512 + g * GSTR);
#pragma unroll
                for (int g = 0; g < 4; ++g) acc[g] = wmma_bf(aA, bA[g], acc[g]);
                aA = fragA(arow + (kc + 2) * 32);
#pragma unroll
                for (int g = 0; g < 4; ++g) bA[g] = fragB(wb0 + (size_t)(kc + 2) * 512 + g * GSTR);
#pragma unroll
                for (int g = 0; g < 4; ++g) acc[g] = wmma_bf(aB, bB[g], acc[g]);
            }
            aB = fragA(arow + (KCN - 1) * 32);
#pragma unroll
            for (int g = 0; g < 4; ++g) bB[g] = fragB(wb0 + (size_t)(KCN - 1) * 512 + g * GSTR);
#pragma unroll
            for (int g = 0; g < 4; ++g) acc[g] = wmma_bf(aA, bA[g], acc[g]);
#pragma unroll
            for (int g = 0; g < 4; ++g) acc[g] = wmma_bf(aB, bB[g], acc[g]);

#pragma unroll
            for (int v = 0; v < 8; ++v) {
                float i_ = sigf(acc[0][v]);
                float f_ = sigf(acc[1][v]);
                float g_ = tanhff(acc[2][v]);
                float o_ = sigf(acc[3][v]);
                float cn = f_ * c0reg[v] + i_ * g_;
                float hn = o_ * tanhff(cn);
                c0reg[v] = cn;
                h0W[(size_t)(rt * 16 + v + 8 * hi) * Hs + ut * 16 + nn] = (__bf16)hn;
            }
            // warm next phase's cold streams while waiting at the barrier
            __builtin_prefetch((const void*)wb1i, 0, 1);
            __builtin_prefetch((const void*)wb1h, 0, 1);
            __builtin_prefetch((const void*)(h1R + (size_t)(rt * 16 + nn) * Hs + hi * 8), 0, 1);
        }
        gbar(bar);

        // ----------------- layer 1: g = h0new@Wih1^T + h1@Whh1^T + b1 -----------------
        {
            v8f acc[4];
#pragma unroll
            for (int g = 0; g < 4; ++g) {
#pragma unroll
                for (int v = 0; v < 8; ++v) acc[g][v] = bb1[g];
            }
            const __bf16* arow0 = h0W + (size_t)(rt * 16 + nn) * Hs + hi * 8;
            const __bf16* arow1 = h1R + (size_t)(rt * 16 + nn) * Hs + hi * 8;
            v16bf a0A = fragA(arow0), a1A = fragA(arow1);
            v16bf biA[4], bhA[4], biB[4], bhB[4];
            v16bf a0B, a1B;
#pragma unroll
            for (int g = 0; g < 4; ++g) {
                biA[g] = fragB(wb1i + g * GSTR);
                bhA[g] = fragB(wb1h + g * GSTR);
            }
#pragma unroll 1
            for (int kc = 0; kc < KCN - 2; kc += 2) {
                const size_t k1 = (size_t)(kc + 1) * 512, k2 = (size_t)(kc + 2) * 512;
                a0B = fragA(arow0 + (kc + 1) * 32);
                a1B = fragA(arow1 + (kc + 1) * 32);
#pragma unroll
                for (int g = 0; g < 4; ++g) {
                    biB[g] = fragB(wb1i + k1 + g * GSTR);
                    bhB[g] = fragB(wb1h + k1 + g * GSTR);
                }
#pragma unroll
                for (int g = 0; g < 4; ++g) {
                    acc[g] = wmma_bf(a0A, biA[g], acc[g]);
                    acc[g] = wmma_bf(a1A, bhA[g], acc[g]);
                }
                a0A = fragA(arow0 + (kc + 2) * 32);
                a1A = fragA(arow1 + (kc + 2) * 32);
#pragma unroll
                for (int g = 0; g < 4; ++g) {
                    biA[g] = fragB(wb1i + k2 + g * GSTR);
                    bhA[g] = fragB(wb1h + k2 + g * GSTR);
                }
#pragma unroll
                for (int g = 0; g < 4; ++g) {
                    acc[g] = wmma_bf(a0B, biB[g], acc[g]);
                    acc[g] = wmma_bf(a1B, bhB[g], acc[g]);
                }
            }
            {
                const size_t kl = (size_t)(KCN - 1) * 512;
                a0B = fragA(arow0 + (KCN - 1) * 32);
                a1B = fragA(arow1 + (KCN - 1) * 32);
#pragma unroll
                for (int g = 0; g < 4; ++g) {
                    biB[g] = fragB(wb1i + kl + g * GSTR);
                    bhB[g] = fragB(wb1h + kl + g * GSTR);
                }
#pragma unroll
                for (int g = 0; g < 4; ++g) {
                    acc[g] = wmma_bf(a0A, biA[g], acc[g]);
                    acc[g] = wmma_bf(a1A, bhA[g], acc[g]);
                }
#pragma unroll
                for (int g = 0; g < 4; ++g) {
                    acc[g] = wmma_bf(a0B, biB[g], acc[g]);
                    acc[g] = wmma_bf(a1B, bhB[g], acc[g]);
                }
            }

#pragma unroll
            for (int v = 0; v < 8; ++v) {
                float i_ = sigf(acc[0][v]);
                float f_ = sigf(acc[1][v]);
                float g_ = tanhff(acc[2][v]);
                float o_ = sigf(acc[3][v]);
                float cn = f_ * c1reg[v] + i_ * g_;
                float hn = o_ * tanhff(cn);
                c1reg[v] = cn;
                h1W[(size_t)(rt * 16 + v + 8 * hi) * Hs + ut * 16 + nn] = (__bf16)hn;
            }
            // warm next phase's / next step's cold stream heads
            __builtin_prefetch((const void*)wb0, 0, 1);
            if (blockIdx.x == 0 && wv < 4)
                __builtin_prefetch((const void*)wb2, 0, 1);
        }
        gbar(bar);

        // ------- layer 2, fused inside block 0 (LDS + __syncthreads only) -------
        if (blockIdx.x == 0) {
            // GEMM g2 = h1new @ Wih2^T  (64x1024 @ 1024x32): wave wv (<4) takes
            // batch tile rt2 = wv and BOTH 16-col gate tiles.
            if (wv < 4) {
                v8f acc0 = {0,0,0,0,0,0,0,0};
                v8f acc1 = {0,0,0,0,0,0,0,0};
                const __bf16* arow = h1W + (size_t)(wv * 16 + nn) * Hs + hi * 8;
                v16bf aA = fragA(arow);
                v16bf b0A = fragB(wb2);
                v16bf b1A = fragB(wb2 + (size_t)KCN * 512);
                v16bf aB, b0B, b1B;
#pragma unroll 1
                for (int kc = 0; kc < KCN - 2; kc += 2) {
                    aB  = fragA(arow + (kc + 1) * 32);
                    b0B = fragB(wb2 + (size_t)(kc + 1) * 512);
                    b1B = fragB(wb2 + (size_t)(KCN + kc + 1) * 512);
                    acc0 = wmma_bf(aA, b0A, acc0);
                    acc1 = wmma_bf(aA, b1A, acc1);
                    aA  = fragA(arow + (kc + 2) * 32);
                    b0A = fragB(wb2 + (size_t)(kc + 2) * 512);
                    b1A = fragB(wb2 + (size_t)(KCN + kc + 2) * 512);
                    acc0 = wmma_bf(aB, b0B, acc0);
                    acc1 = wmma_bf(aB, b1B, acc1);
                }
                aB  = fragA(arow + (KCN - 1) * 32);
                b0B = fragB(wb2 + (size_t)(KCN - 1) * 512);
                b1B = fragB(wb2 + (size_t)(2 * KCN - 1) * 512);
                acc0 = wmma_bf(aA, b0A, acc0);
                acc1 = wmma_bf(aA, b1A, acc1);
                acc0 = wmma_bf(aB, b0B, acc0);
                acc1 = wmma_bf(aB, b1B, acc1);
#pragma unroll
                for (int v = 0; v < 8; ++v) {
                    const int m = wv * 16 + v + 8 * hi;
                    g2s[m * 32 + nn]      = acc0[v];
                    g2s[m * 32 + 16 + nn] = acc1[v];
                }
            }
            __syncthreads();
            // activation + pred write (wave 0)
            if (wv == 0) {
                float hnew[16];
#pragma unroll
                for (int it = 0; it < 16; ++it) {
                    const int idx = it * 32 + lane;       // 0..511
                    const int b = idx >> 3;               // u == u2 == lane&7
                    const float* hdr = hdbuf + b * 8;     // previous hd (reads only)
                    float di = 0.f, df = 0.f, dg = 0.f, dn = 0.f;
#pragma unroll
                    for (int k = 0; k < 8; ++k) {
                        const float hv = hdr[k];
                        di += hv * w2i[k];
                        df += hv * w2f[k];
                        dg += hv * w2g[k];
                        dn += hv * w2o[k];
                    }
                    const float* g2 = g2s + b * 32;
                    float i_ = sigf  (g2[u2]      + b2i + di);
                    float f_ = sigf  (g2[u2 + 8]  + b2f + df);
                    float g_ = tanhff(g2[u2 + 16] + b2g + dg);
                    float o_ = sigf  (g2[u2 + 24] + b2o + dn);
                    float cn = f_ * c2reg[it] + i_ * g_;
                    float hn = o_ * tanhff(cn);
                    c2reg[it] = cn;
                    hnew[it] = hn;
                }
#pragma unroll
                for (int it = 0; it < 16; ++it) {         // all loads done; now store
                    const int idx = it * 32 + lane;
                    hdbuf[idx] = hnew[it];
                    if (t >= ENCs - 1) {
                        const int b = idx >> 3;
                        outp[((size_t)b * HORs + (t - (ENCs - 1))) * Ds + u2] = hnew[it];
                    }
                }
            }
        }
        gbar(bar);
    }
}

__global__ void ycopy_kernel(const float* __restrict__ inp, float* __restrict__ ydst) {
    size_t idx = (size_t)blockIdx.x * blockDim.x + threadIdx.x;
    if (idx >= (size_t)PRED_N) return;
    size_t b   = idx / (HORs * Ds);
    size_t rem = idx % (HORs * Ds);
    ydst[idx] = inp[b * ((ENCs + HORs) * Ds) + ENCs * Ds + rem];
}

__global__ void loss_kernel(const float* __restrict__ out, float* __restrict__ lossdst) {
    const int d = blockIdx.x;   // 0..7
    __shared__ float red[256];
    float s = 0.f;
    for (int i = threadIdx.x; i < Bs * HORs; i += 256) {
        float p  = out[(size_t)i * Ds + d];
        float yy = out[PRED_N + (size_t)i * Ds + d];
        float df = p - yy;
        s += df * df;
    }
    red[threadIdx.x] = s;
    __syncthreads();
    for (int o = 128; o > 0; o >>= 1) {
        if (threadIdx.x < o) red[threadIdx.x] += red[threadIdx.x + o];
        __syncthreads();
    }
    if (threadIdx.x == 0) lossdst[d] = red[0] / (float)(Bs * HORs);
}

extern "C" void kernel_launch(void* const* d_in, const int* in_sizes, int n_in,
                              void* d_out, int out_size, void* d_ws, size_t ws_size,
                              hipStream_t stream) {
    const float* inp  = (const float*)d_in[0];
    const float* wih0 = (const float*)d_in[1];
    const float* whh0 = (const float*)d_in[2];
    const float* b0   = (const float*)d_in[3];
    const float* wih1 = (const float*)d_in[4];
    const float* whh1 = (const float*)d_in[5];
    const float* b1   = (const float*)d_in[6];
    const float* wih2 = (const float*)d_in[7];
    const float* whh2 = (const float*)d_in[8];
    const float* b2   = (const float*)d_in[9];
    float* out = (float*)d_out;

    char* ws = (char*)d_ws;
    size_t off = 0;
    auto walloc = [&](size_t bytes) -> void* {
        void* p = (void*)(ws + off);
        off = (off + bytes + 255u) & ~(size_t)255u;
        return p;
    };
    const size_t BIGPACK = (size_t)256 * KCN * 512;   // elems per 4096x1024 matrix
    __bf16* pWhh0 = (__bf16*)walloc(BIGPACK * 2);
    __bf16* pWih1 = (__bf16*)walloc(BIGPACK * 2);
    __bf16* pWhh1 = (__bf16*)walloc(BIGPACK * 2);
    __bf16* pWih2 = (__bf16*)walloc((size_t)2 * KCN * 512 * 2);
    char* state0 = ws + off;
    __bf16*   h0b   = (__bf16*)walloc((size_t)2 * Bs * Hs * 2);
    __bf16*   h1b   = (__bf16*)walloc((size_t)2 * Bs * Hs * 2);
    float*    hdbuf = (float*)walloc((size_t)Bs * Ds * 4);
    unsigned* bar   = (unsigned*)walloc(256);
    const size_t state_bytes = (size_t)((ws + off) - state0);

    // zero recurrent state + barrier words (capture-safe memset node)
    hipMemsetAsync(state0, 0, state_bytes, stream);

    // pack weights into per-lane bf16 WMMA B-fragments (once per launch)
    const int PT = 256;
    pack_weights<<<(unsigned)((BIGPACK + PT - 1) / PT), PT, 0, stream>>>(whh0, pWhh0, 256, Hs);
    pack_weights<<<(unsigned)((BIGPACK + PT - 1) / PT), PT, 0, stream>>>(wih1, pWih1, 256, Hs);
    pack_weights<<<(unsigned)((BIGPACK + PT - 1) / PT), PT, 0, stream>>>(whh1, pWhh1, 256, Hs);
    pack_weights<<<(unsigned)(((size_t)2 * KCN * 512 + PT - 1) / PT), PT, 0, stream>>>(wih2, pWih2, 2, Hs);

    // full 431-step recurrence in one persistent kernel
    lstm_persistent<<<NBLK, TPB, 0, stream>>>(
        inp, wih0, b0, b1, whh2, b2,
        pWhh0, pWih1, pWhh1, pWih2,
        h0b, h1b, hdbuf, bar, out);

    // y = inputs[:, ENC:, :] ; loss = mean((pred-y)^2, axis=(0,1))
    ycopy_kernel<<<(PRED_N + 255) / 256, 256, 0, stream>>>(inp, out + PRED_N);
    loss_kernel<<<Ds, 256, 0, stream>>>(out, out + 2 * (size_t)PRED_N);
}
